// SequenceClassification_9852654977315
// MI455X (gfx1250) — matrix operand order repaired and checked by reading, not compiled, not run on hardware
//
#include <hip/hip_runtime.h>
#include <math.h>

// ---------------- model dims ----------------
#define NLAYER 4
#define HD     768
#define NH     12
#define DH     64
#define WWIN   256
#define SEQ    4096
#define BATCH  2
#define FF     3072
#define NLCLS  3
#define MTOK   (BATCH*SEQ)   // 8192

typedef __attribute__((ext_vector_type(16))) _Float16 v16h;
typedef __attribute__((ext_vector_type(8)))  _Float16 h8;
typedef __attribute__((ext_vector_type(8)))  float    v8f;
typedef __attribute__((ext_vector_type(4)))  unsigned int u32x4;
typedef __attribute__((ext_vector_type(8)))  int          i32x8;
typedef __attribute__((ext_vector_type(4)))  int          i32x4;

// ---------------------------------------------------------------------------
// weight convert fp32 [K,N] -> f16 transposed [N,K]
// ---------------------------------------------------------------------------
__global__ void cvt_w_T(const float* __restrict__ w, _Float16* __restrict__ wt,
                        int K, int N) {
  long i = (long)blockIdx.x * blockDim.x + threadIdx.x;
  if (i < (long)K * N) {
    int k = (int)(i / N), n = (int)(i % N);
    wt[(long)n * K + k] = (_Float16)w[i];
  }
}

// ---------------------------------------------------------------------------
// block reduction helper (blockDim.x == 256)
// ---------------------------------------------------------------------------
__device__ __forceinline__ float block_sum(float v, float* sbuf) {
  int t = threadIdx.x;
  sbuf[t] = v; __syncthreads();
  #pragma unroll
  for (int s = 128; s > 0; s >>= 1) {
    if (t < s) sbuf[t] += sbuf[t + s];
    __syncthreads();
  }
  float r = sbuf[0]; __syncthreads();
  return r;
}

// ---------------------------------------------------------------------------
// embedding gather + layernorm -> x (f32) and x16 (f16)
// ---------------------------------------------------------------------------
__global__ __launch_bounds__(256)
void embed_ln(const int* __restrict__ ids, const float* __restrict__ tok,
              const float* __restrict__ pos, const float* __restrict__ g,
              const float* __restrict__ be, float* __restrict__ x,
              _Float16* __restrict__ x16) {
  __shared__ float sbuf[256];
  const int t = blockIdx.x;            // token index 0..MTOK-1
  const int sIdx = t % SEQ;
  const int id = ids[t];
  float vals[3], sum = 0.f;
  #pragma unroll
  for (int i = 0; i < 3; ++i) {
    int d = threadIdx.x + i * 256;
    float v = tok[(long)id * HD + d] + pos[(long)sIdx * HD + d];
    vals[i] = v; sum += v;
  }
  float mean = block_sum(sum, sbuf) * (1.f / HD);
  float var = 0.f;
  #pragma unroll
  for (int i = 0; i < 3; ++i) { float dv = vals[i] - mean; var += dv * dv; }
  var = block_sum(var, sbuf) * (1.f / HD);
  float rstd = rsqrtf(var + 1e-5f);
  #pragma unroll
  for (int i = 0; i < 3; ++i) {
    int d = threadIdx.x + i * 256;
    float o = (vals[i] - mean) * rstd * g[d] + be[d];
    x[(long)t * HD + d] = o;
    x16[(long)t * HD + d] = (_Float16)o;
  }
}

// ---------------------------------------------------------------------------
// residual add + layernorm: x = LN(x + tmp)*g + b  -> x (f32) and x16 (f16)
// ---------------------------------------------------------------------------
__global__ __launch_bounds__(256)
void add_ln(const float* __restrict__ xin, const float* __restrict__ tm,
            const float* __restrict__ g, const float* __restrict__ be,
            float* __restrict__ xout, _Float16* __restrict__ x16) {
  __shared__ float sbuf[256];
  const int t = blockIdx.x;
  float vals[3], sum = 0.f;
  #pragma unroll
  for (int i = 0; i < 3; ++i) {
    int d = threadIdx.x + i * 256;
    float v = xin[(long)t * HD + d] + tm[(long)t * HD + d];
    vals[i] = v; sum += v;
  }
  float mean = block_sum(sum, sbuf) * (1.f / HD);
  float var = 0.f;
  #pragma unroll
  for (int i = 0; i < 3; ++i) { float dv = vals[i] - mean; var += dv * dv; }
  var = block_sum(var, sbuf) * (1.f / HD);
  float rstd = rsqrtf(var + 1e-5f);
  #pragma unroll
  for (int i = 0; i < 3; ++i) {
    int d = threadIdx.x + i * 256;
    float o = (vals[i] - mean) * rstd * g[d] + be[d];
    xout[(long)t * HD + d] = o;
    x16[(long)t * HD + d] = (_Float16)o;
  }
}

// ---------------------------------------------------------------------------
// WMMA GEMM with TDM-staged B tiles:
//   C[M,N] = A[M,K](f16) @ BT[N,K](f16)^T + bias
//   EPI 0: f16 out (q/k/v)   EPI 1: f32 out (residual path)   EPI 2: GELU f16
// block = 128 threads (4 waves); block tile 64x64; wave -> 16x64 strip.
// B tile (64 N-rows x 32 K) is DMA'd into LDS by the Tensor Data Mover,
// double buffered, tracked on TENSORcnt. TDM fill order [n][k] matches the
// B-fragment layout (lanes 0-15: K 0..7,16..23, lanes 16-31: K 8..15,24..31, N = lane&15).
// ---------------------------------------------------------------------------
template<int EPI>
__global__ __launch_bounds__(128)
void gemm_wmma(const _Float16* __restrict__ A, const _Float16* __restrict__ BT,
               const float* __restrict__ bias, float* __restrict__ Cf,
               _Float16* __restrict__ Ch, int Mdim, int Ndim, int Kdim) {
  __shared__ __align__(16) _Float16 ldsB[2][64 * 32];
  (void)Mdim;

  const int lane = threadIdx.x & 31;
  const int wv   = threadIdx.x >> 5;
  const int l16  = lane & 15;
  const int hlf  = lane >> 4;
  const int row0 = blockIdx.y * 64 + wv * 16;
  const int col0 = blockIdx.x * 64;
  const int T    = Kdim >> 5;            // number of 32-wide K steps

  // ---- TDM descriptor issue: 2D tile, tile_dim0=32 (K, contiguous),
  //      tile_dim1=64 (N rows), stride = Kdim elements, data_size = 2B ----
  auto issue_tile = [&](int buf, int k0) {
    unsigned long long ga =
        (unsigned long long)(const void*)(BT + (long)col0 * Kdim + k0);
    unsigned ldsa = (unsigned)(size_t)(&ldsB[buf][0]);
    u32x4 g0;
    g0[0] = 1u;                                          // count=1, user mode
    g0[1] = ldsa;                                        // lds_addr [63:32]
    g0[2] = (unsigned)(ga & 0xffffffffu);                // global_addr lo
    g0[3] = (unsigned)((ga >> 32) & 0x01ffffffu)         // global_addr [56:32]
          | (2u << 30);                                  // type = 2 (image)
    i32x8 g1;
    g1[0] = (int)(1u << 16);                             // data_size=1 -> 2B
    g1[1] = (int)(((unsigned)Kdim & 0xffffu) << 16);     // tensor_dim0 lo16
    g1[2] = (int)((((unsigned)Kdim >> 16) & 0xffffu)     // tensor_dim0 hi16
          | (((unsigned)Ndim & 0xffffu) << 16));         // tensor_dim1 lo16
    g1[3] = (int)((((unsigned)Ndim >> 16) & 0xffffu)     // tensor_dim1 hi16
          | (32u << 16));                                // tile_dim0 = 32
    g1[4] = 64;                                          // tile_dim1 = 64
    g1[5] = (int)(unsigned)Kdim;                         // dim0_stride lo32
    g1[6] = 0;                                           // dim0_stride hi16
    g1[7] = 0;
    i32x4 z4 = {0, 0, 0, 0};
#if __clang_major__ >= 23
    i32x8 z8 = {0, 0, 0, 0, 0, 0, 0, 0};
    __builtin_amdgcn_tensor_load_to_lds(g0, g1, z4, z4, z8, 0);
#else
    __builtin_amdgcn_tensor_load_to_lds(g0, g1, z4, z4, 0);
#endif
  };

  if (wv == 0) issue_tile(0, 0);

  v8f acc0 = {}, acc1 = {}, acc2 = {}, acc3 = {};
  const long arow = (long)(row0 + l16) * Kdim;
  const int kbA = hlf * 8;

  for (int t = 0; t < T; ++t) {
    const int cur = t & 1;
    if (wv == 0) {
      if (t + 1 < T) {
        issue_tile(cur ^ 1, (t + 1) << 5);
        __builtin_amdgcn_s_wait_tensorcnt((short)1);   // current tile done
      } else {
        __builtin_amdgcn_s_wait_tensorcnt((short)0);
      }
    }
    __syncthreads();                                   // buf[cur] visible

    // A fragment: direct global vector loads (rows unique per wave)
    const _Float16* ap = A + arow + (t << 5) + kbA;
    h8 alo = *(const h8*)ap;
    h8 ahi = *(const h8*)(ap + 16);
    v16h af;
    #pragma unroll
    for (int i = 0; i < 8; ++i) { af[i] = alo[i]; af[8 + i] = ahi[i]; }

    #pragma unroll
    for (int nc = 0; nc < 4; ++nc) {
      const _Float16* bp = &ldsB[cur][(nc * 16 + l16) * 32 + hlf * 8];
      h8 b0 = *(const h8*)bp;
      h8 b1 = *(const h8*)(bp + 16);
      v16h bf;
      #pragma unroll
      for (int i = 0; i < 8; ++i) { bf[i] = b0[i]; bf[8 + i] = b1[i]; }
      v8f& acc = nc == 0 ? acc0 : nc == 1 ? acc1 : nc == 2 ? acc2 : acc3;
      acc = __builtin_amdgcn_wmma_f32_16x16x32_f16(false, af, false, bf,
                                                   (short)0, acc, false, false);
    }
    __syncthreads();                 // all reads done before buf reuse
  }

  #pragma unroll
  for (int nc = 0; nc < 4; ++nc) {
    v8f acc = nc == 0 ? acc0 : nc == 1 ? acc1 : nc == 2 ? acc2 : acc3;
    const int n = col0 + nc * 16 + l16;
    const float bsv = bias[n];
    #pragma unroll
    for (int v = 0; v < 8; ++v) {
      const int r = row0 + v + hlf * 8;
      float val = acc[v] + bsv;
      if (EPI == 2) val = 0.5f * val * (1.0f + erff(val * 0.70710678118654752f));
      if (EPI == 1) Cf[(long)r * Ndim + n] = val;
      else          Ch[(long)r * Ndim + n] = (_Float16)val;
    }
  }
}

// ---------------------------------------------------------------------------
// Banded (sliding-window) attention, flash-style online softmax.
// grid.x = B*NH*(SEQ/WWIN)*(WWIN/16); one wave per 16-query tile.
// q/k/v/out layout: [B, S, NH, DH] (token-major, heads contiguous = HD).
// ---------------------------------------------------------------------------
__global__ __launch_bounds__(32)
void attn_swa(const _Float16* __restrict__ q16, const _Float16* __restrict__ k16,
              const _Float16* __restrict__ v16, _Float16* __restrict__ out16) {
  __shared__ __align__(16) _Float16 ldsP[16 * 32];
  __shared__ __align__(16) _Float16 ldsV[32 * 80];   // stride 80 halves (16B mult)

  int idx = blockIdx.x;
  const int qt = idx & 15; idx >>= 4;     // WWIN/16 = 16 subtiles
  const int c  = idx & 15; idx >>= 4;     // SEQ/WWIN = 16 chunks
  const int h  = idx % NH;
  const int b  = idx / NH;

  const int lane = threadIdx.x & 31;
  const int l16  = lane & 15;
  const int hlf  = lane >> 4;
  const int kb   = hlf * 8;

  // --- Q A-fragments (K = head dim 0..31 and 32..63) ---
  const int qrowA = c * WWIN + qt * 16 + l16;
  const _Float16* qp = q16 + (((long)b * SEQ + qrowA) * NH + h) * DH;
  v16h qa0, qa1;
  {
    h8 x0 = *(const h8*)(qp + kb);
    h8 x1 = *(const h8*)(qp + 16 + kb);
    h8 x2 = *(const h8*)(qp + 32 + kb);
    h8 x3 = *(const h8*)(qp + 48 + kb);
    #pragma unroll
    for (int i = 0; i < 8; ++i) {
      qa0[i] = x0[i]; qa0[8 + i] = x1[i];
      qa1[i] = x2[i]; qa1[8 + i] = x3[i];
    }
  }

  float mrun[8], lrun[8];
  #pragma unroll
  for (int v = 0; v < 8; ++v) { mrun[v] = -1e30f; lrun[v] = 0.f; }
  v8f o0 = {}, o1 = {}, o2 = {}, o3 = {};

  const int key0base = c * WWIN - WWIN;   // global pos of column 0 of 3w block

  for (int j = 0; j < 24; ++j) {          // 3*WWIN / 32 key tiles
    // ---- scores: two 16x16 tiles (keys j*32+0..15 and +16..31) ----
    v8f s0t, s1t;
    #pragma unroll
    for (int t = 0; t < 2; ++t) {
      int kg = key0base + j * 32 + t * 16 + l16;
      int kgc = kg < 0 ? 0 : (kg >= SEQ ? SEQ - 1 : kg);
      const _Float16* kp = k16 + (((long)b * SEQ + kgc) * NH + h) * DH + hlf * 8;
      h8 ka = *(const h8*)kp;
      h8 kbv = *(const h8*)(kp + 16);
      h8 kc = *(const h8*)(kp + 32);
      h8 kd = *(const h8*)(kp + 48);
      v16h bf0, bf1;
      #pragma unroll
      for (int i = 0; i < 8; ++i) {
        bf0[i] = ka[i]; bf0[8 + i] = kbv[i];
        bf1[i] = kc[i]; bf1[8 + i] = kd[i];
      }
      v8f sc = {};
      sc = __builtin_amdgcn_wmma_f32_16x16x32_f16(false, qa0, false, bf0,
                                                  (short)0, sc, false, false);
      sc = __builtin_amdgcn_wmma_f32_16x16x32_f16(false, qa1, false, bf1,
                                                  (short)0, sc, false, false);
      if (t == 0) s0t = sc; else s1t = sc;
    }

    // ---- mask + scale + online softmax update ----
    const int kg0 = key0base + j * 32 + l16;
    const int kg1 = kg0 + 16;
    float p0a[8], p1a[8];
    #pragma unroll
    for (int v = 0; v < 8; ++v) {
      const int qg = c * WWIN + qt * 16 + v + hlf * 8;
      int d0 = kg0 - qg; if (d0 < 0) d0 = -d0;
      int d1 = kg1 - qg; if (d1 < 0) d1 = -d1;
      bool ok0 = (kg0 >= 0) && (kg0 < SEQ) && (d0 <= WWIN);
      bool ok1 = (kg1 >= 0) && (kg1 < SEQ) && (d1 <= WWIN);
      float sv0 = ok0 ? s0t[v] * 0.125f : -1e30f;
      float sv1 = ok1 ? s1t[v] * 0.125f : -1e30f;
      float tm = fmaxf(sv0, sv1);
      tm = fmaxf(tm, __shfl_xor(tm, 1, 32));
      tm = fmaxf(tm, __shfl_xor(tm, 2, 32));
      tm = fmaxf(tm, __shfl_xor(tm, 4, 32));
      tm = fmaxf(tm, __shfl_xor(tm, 8, 32));
      float mn = fmaxf(mrun[v], tm);
      float al = __expf(mrun[v] - mn);
      float p0 = __expf(sv0 - mn);
      float p1 = __expf(sv1 - mn);
      float rs = p0 + p1;
      rs += __shfl_xor(rs, 1, 32);
      rs += __shfl_xor(rs, 2, 32);
      rs += __shfl_xor(rs, 4, 32);
      rs += __shfl_xor(rs, 8, 32);
      lrun[v] = lrun[v] * al + rs;
      mrun[v] = mn;
      o0[v] *= al; o1[v] *= al; o2[v] *= al; o3[v] *= al;
      p0a[v] = p0; p1a[v] = p1;
    }

    // ---- stage P (16x32 f16) and V tile (32x64 f16) in LDS ----
    #pragma unroll
    for (int v = 0; v < 8; ++v) {
      int r = v + hlf * 8;
      ldsP[r * 32 + l16]      = (_Float16)p0a[v];
      ldsP[r * 32 + 16 + l16] = (_Float16)p1a[v];
    }
    {
      int kg = key0base + j * 32 + lane;
      int kgc = kg < 0 ? 0 : (kg >= SEQ ? SEQ - 1 : kg);
      bool ok = (kg >= 0) && (kg < SEQ);
      const _Float16* vp = v16 + (((long)b * SEQ + kgc) * NH + h) * DH;
      h8 zz = {};
      #pragma unroll
      for (int q8 = 0; q8 < 8; ++q8) {
        h8 vv = *(const h8*)(vp + q8 * 8);
        vv = ok ? vv : zz;
        *(h8*)(&ldsV[lane * 80 + q8 * 8]) = vv;
      }
    }
    __syncthreads();

    // ---- P A-fragment from LDS ----
    v16h pa;
    {
      const _Float16* pp = &ldsP[l16 * 32 + kb];
      h8 pl = *(const h8*)pp;
      h8 ph = *(const h8*)(pp + 16);
      #pragma unroll
      for (int i = 0; i < 8; ++i) { pa[i] = pl[i]; pa[8 + i] = ph[i]; }
    }
    // ---- O += P @ V (4 dim chunks) ----
    #pragma unroll
    for (int dc = 0; dc < 4; ++dc) {
      v16h vb;
      #pragma unroll
      for (int i = 0; i < 16; ++i)
        vb[i] = ldsV[(hlf * 8 + (i & 7) + (i >> 3) * 16) * 80 + dc * 16 + l16];
      v8f& oo = dc == 0 ? o0 : dc == 1 ? o1 : dc == 2 ? o2 : o3;
      oo = __builtin_amdgcn_wmma_f32_16x16x32_f16(false, pa, false, vb,
                                                  (short)0, oo, false, false);
    }
    __syncthreads();
  }

  // ---- normalize and store [B,S,NH,DH] f16 ----
  #pragma unroll
  for (int dc = 0; dc < 4; ++dc) {
    v8f oo = dc == 0 ? o0 : dc == 1 ? o1 : dc == 2 ? o2 : o3;
    #pragma unroll
    for (int v = 0; v < 8; ++v) {
      int row = v + hlf * 8;
      int qg = c * WWIN + qt * 16 + row;
      float inv = lrun[v] > 0.f ? 1.f / lrun[v] : 0.f;
      out16[(((long)b * SEQ + qg) * NH + h) * DH + dc * 16 + l16] =
          (_Float16)(oo[v] * inv);
    }
  }
}

// ---------------------------------------------------------------------------
// mean pool over sequence, then tiny classifier
// ---------------------------------------------------------------------------
__global__ __launch_bounds__(256)
void pool_mean(const float* __restrict__ x, float* __restrict__ pooled) {
  __shared__ float sbuf[256];
  int bd = blockIdx.x;                 // 0 .. B*HD-1
  int b = bd / HD, d = bd % HD;
  float sum = 0.f;
  for (int s = threadIdx.x; s < SEQ; s += 256)
    sum += x[((long)b * SEQ + s) * HD + d];
  sum = block_sum(sum, sbuf);
  if (threadIdx.x == 0) pooled[bd] = sum * (1.f / SEQ);
}

__global__ void cls_head(const float* __restrict__ pooled,
                         const float* __restrict__ w,
                         const float* __restrict__ bias,
                         float* __restrict__ out) {
  int t = threadIdx.x;
  if (t < BATCH * NLCLS) {
    int b = t / NLCLS, c = t % NLCLS;
    float acc = bias[c];
    for (int d = 0; d < HD; ++d) acc += pooled[b * HD + d] * w[d * NLCLS + c];
    out[t] = acc;
  }
}

// ---------------------------------------------------------------------------
extern "C" void kernel_launch(void* const* d_in, const int* in_sizes, int n_in,
                              void* d_out, int out_size, void* d_ws, size_t ws_size,
                              hipStream_t stream) {
  (void)in_sizes; (void)n_in; (void)out_size; (void)ws_size;
  const int*   ids   = (const int*)  d_in[0];
  const float* tok   = (const float*)d_in[1];
  const float* pos   = (const float*)d_in[2];
  const float* eln_s = (const float*)d_in[3];
  const float* eln_b = (const float*)d_in[4];
  const float* Wq    = (const float*)d_in[5];
  const float* bq    = (const float*)d_in[6];
  const float* Wk    = (const float*)d_in[7];
  const float* bk    = (const float*)d_in[8];
  const float* Wv    = (const float*)d_in[9];
  const float* bv    = (const float*)d_in[10];
  const float* Wo    = (const float*)d_in[11];
  const float* bo    = (const float*)d_in[12];
  const float* ln1s  = (const float*)d_in[13];
  const float* ln1b  = (const float*)d_in[14];
  const float* W1    = (const float*)d_in[15];
  const float* b1    = (const float*)d_in[16];
  const float* W2    = (const float*)d_in[17];
  const float* b2    = (const float*)d_in[18];
  const float* ln2s  = (const float*)d_in[19];
  const float* ln2b  = (const float*)d_in[20];
  const float* clsw  = (const float*)d_in[21];
  const float* clsb  = (const float*)d_in[22];
  float* out = (float*)d_out;

  // ---- scratch layout ----
  char* ws = (char*)d_ws;
  size_t off = 0;
  auto carve = [&](size_t bytes) -> void* {
    void* p = ws + off;
    off += (bytes + 255) & ~(size_t)255;
    return p;
  };
  float*     x      = (float*)    carve((size_t)MTOK * HD * 4);
  _Float16*  x16    = (_Float16*) carve((size_t)MTOK * HD * 2);
  _Float16*  q16    = (_Float16*) carve((size_t)MTOK * HD * 2);
  _Float16*  k16    = (_Float16*) carve((size_t)MTOK * HD * 2);
  _Float16*  v16    = (_Float16*) carve((size_t)MTOK * HD * 2);
  _Float16*  a16    = (_Float16*) carve((size_t)MTOK * HD * 2);
  _Float16*  h16    = (_Float16*) carve((size_t)MTOK * FF * 2);
  float*     tmp    = (float*)    carve((size_t)MTOK * HD * 4);
  _Float16*  wT     = (_Float16*) carve((size_t)HD * FF * 2);
  float*     pooled = (float*)    carve((size_t)BATCH * HD * 4);

  // ---- embedding + LN ----
  embed_ln<<<MTOK, 256, 0, stream>>>(ids, tok, pos, eln_s, eln_b, x, x16);

  const dim3 gP(HD / 64, MTOK / 64);    // 12 x 128
  const dim3 gF1(FF / 64, MTOK / 64);   // 48 x 128
  const int cvt1 = (HD * HD + 255) / 256;
  const int cvt2 = (HD * FF + 255) / 256;
  const int attnBlocks = BATCH * NH * (SEQ / WWIN) * (WWIN / 16);  // 6144

  for (int l = 0; l < NLAYER; ++l) {
    // Q / K / V projections
    cvt_w_T<<<cvt1, 256, 0, stream>>>(Wq + (long)l * HD * HD, wT, HD, HD);
    gemm_wmma<0><<<gP, 128, 0, stream>>>(x16, wT, bq + l * HD, nullptr, q16,
                                         MTOK, HD, HD);
    cvt_w_T<<<cvt1, 256, 0, stream>>>(Wk + (long)l * HD * HD, wT, HD, HD);
    gemm_wmma<0><<<gP, 128, 0, stream>>>(x16, wT, bk + l * HD, nullptr, k16,
                                         MTOK, HD, HD);
    cvt_w_T<<<cvt1, 256, 0, stream>>>(Wv + (long)l * HD * HD, wT, HD, HD);
    gemm_wmma<0><<<gP, 128, 0, stream>>>(x16, wT, bv + l * HD, nullptr, v16,
                                         MTOK, HD, HD);
    // banded attention
    attn_swa<<<attnBlocks, 32, 0, stream>>>(q16, k16, v16, a16);
    // output projection (f32 out for residual)
    cvt_w_T<<<cvt1, 256, 0, stream>>>(Wo + (long)l * HD * HD, wT, HD, HD);
    gemm_wmma<1><<<gP, 128, 0, stream>>>(a16, wT, bo + l * HD, tmp, nullptr,
                                         MTOK, HD, HD);
    add_ln<<<MTOK, 256, 0, stream>>>(x, tmp, ln1s + l * HD, ln1b + l * HD,
                                     x, x16);
    // FFN
    cvt_w_T<<<cvt2, 256, 0, stream>>>(W1 + (long)l * HD * FF, wT, HD, FF);
    gemm_wmma<2><<<gF1, 128, 0, stream>>>(x16, wT, b1 + l * FF, nullptr, h16,
                                          MTOK, FF, HD);
    cvt_w_T<<<cvt2, 256, 0, stream>>>(W2 + (long)l * FF * HD, wT, FF, HD);
    gemm_wmma<1><<<gP, 128, 0, stream>>>(h16, wT, b2 + l * HD, tmp, nullptr,
                                         MTOK, HD, FF);
    add_ln<<<MTOK, 256, 0, stream>>>(x, tmp, ln2s + l * HD, ln2b + l * HD,
                                     x, x16);
  }

  pool_mean<<<BATCH * HD, 256, 0, stream>>>(x, pooled);
  cls_head<<<1, 32, 0, stream>>>(pooled, clsw, clsb, out);
}
